// TemporalExtGCN_38628935860927
// MI455X (gfx1250) — compile-verified, matrix operand order checked by reading, not run on hardware
//
#include <hip/hip_runtime.h>

typedef __attribute__((ext_vector_type(16))) _Float16 v16h;
typedef __attribute__((ext_vector_type(8)))  float    v8f;

#define N_NODES 100000
#define N_EDGES 1000000
#define N_REL   7
#define KTOT    512                       // (1 + N_REL) * 64
#define SUMS_FLOATS (N_NODES * N_REL * 64)  // 44,800,000 floats (179.2 MB)
#define CNT_FLOATS  (N_NODES * N_REL)       //    700,000 floats (  2.8 MB)

__device__ __forceinline__ void atomAddF(float* p, float v) {
  // relaxed agent-scope fadd -> expect global_atomic_add_f32 (no CAS loop)
  __hip_atomic_fetch_add(p, v, __ATOMIC_RELAXED, __HIP_MEMORY_SCOPE_AGENT);
}

// -------- zero the sums/counts region of workspace --------
__global__ void rgcn_zero(float* __restrict__ ws, long n) {
  long i = (long)blockIdx.x * blockDim.x + threadIdx.x;
  long stride = (long)gridDim.x * blockDim.x;
  for (; i < n; i += stride) ws[i] = 0.0f;
}

// -------- pack stacked B = [root; W_0..W_6] (512x64 f32) into f16 WMMA
// -------- B-fragment layout: Bp[(chunk*4+ntile)*32 + lane][16 halves] --------
__global__ void rgcn_packB(const float* __restrict__ W,
                           const float* __restrict__ root,
                           _Float16* __restrict__ Bp) {
  const int lane  = threadIdx.x;   // 0..31
  const int ntile = blockIdx.x;    // 0..3
  const int chunk = blockIdx.y;    // 0..15
  const int n = ntile * 16 + (lane & 15);
  const int b = lane >> 4;
  v16h frag;
#pragma unroll
  for (int h = 0; h < 16; ++h) {
    // 16-bit B fragment K mapping (mirrors A layout, ISA 7.12.2)
    const int kk = (h < 8) ? (8 * b + h) : (16 + 8 * b + (h - 8));
    const int k  = chunk * 32 + kk;
    const float v = (k < 64) ? root[k * 64 + n] : W[(k - 64) * 64 + n];
    frag[h] = (_Float16)v;
  }
  *(v16h*)(Bp + (((chunk * 4 + ntile) * 32 + lane) * 16)) = frag;
}

// -------- one wave per edge: scatter x[src] into sums[(dst,rel)], bump count --------
__global__ void rgcn_scatter(const float* __restrict__ x,
                             const int* __restrict__ ei,
                             const int* __restrict__ et,
                             float* __restrict__ sums,
                             float* __restrict__ cnt) {
  const int wave = (blockIdx.x * blockDim.x + threadIdx.x) >> 5;
  const int lane = threadIdx.x & 31;
  const int src = ei[wave];
  const int dst = ei[N_EDGES + wave];
  const int rel = et[wave];
  const float* xr = x + (long)src * 64;
  float* sr = sums + ((long)dst * N_REL + rel) * 64;
  atomAddF(sr + lane,      xr[lane]);
  atomAddF(sr + lane + 32, xr[lane + 32]);
  if (lane == 0) atomAddF(cnt + (long)dst * N_REL + rel, 1.0f);
}

// -------- fused GEMM: out[n] = [x | mean_r] (64x512) @ Bpacked (512x64) + bias --------
__global__ void __launch_bounds__(256)
rgcn_gemm(const float* __restrict__ x,
          const float* __restrict__ bias,
          const float* __restrict__ sums,
          const float* __restrict__ cnt,
          const _Float16* __restrict__ Bp,
          float* __restrict__ out) {
  // A fragments for 4 M-tiles x 16 K-chunks, pre-packed per-lane: 64 KB
  __shared__ __align__(32) _Float16 sA[4 * 16 * 32 * 16];
  const int tid   = threadIdx.x;
  const int node0 = blockIdx.x * 64;

  // ---- stage A into packed fragment layout (f32 -> f16, mean on the fly) ----
  for (int idx = tid; idx < 64 * KTOT; idx += 256) {
    const int nl = idx >> 9;        // local node 0..63
    const int k  = idx & 511;       // K index 0..511
    const int node = node0 + nl;
    float v = 0.0f;
    if (node < N_NODES) {
      if (k < 64) {
        v = x[(long)node * 64 + k];
      } else {
        const int r = (k >> 6) - 1;
        const int d = k & 63;
        const long sg = (long)node * N_REL + r;
        v = sums[sg * 64 + d] / fmaxf(cnt[sg], 1.0f);
      }
    }
    // pack to per-lane 16-bit A-fragment layout (ISA 7.12.2)
    const int mtile = nl >> 4, m = nl & 15;
    const int chunk = k >> 5,  kk = k & 31;
    const int b = (kk >> 3) & 1;
    const int h = (kk & 7) + ((kk >> 4) << 3);
    const int lane = m + (b << 4);
    sA[(((mtile * 16 + chunk) * 32) + lane) * 16 + h] = (_Float16)v;
  }
  __syncthreads();

  // ---- WMMA: wave w -> M-tile w/2, N-tiles {2*(w&1), 2*(w&1)+1} ----
  const int w     = tid >> 5;
  const int lane  = tid & 31;
  const int mtile = w >> 1;
  const int nt0   = (w & 1) * 2;
  v8f c0 = {}; v8f c1 = {};
#pragma unroll
  for (int chunk = 0; chunk < 16; ++chunk) {
    const v16h a  = *(const v16h*)&sA[((mtile * 16 + chunk) * 32 + lane) * 16];
    const v16h b0 = *(const v16h*)(Bp + ((chunk * 4 + nt0)     * 32 + lane) * 16);
    const v16h b1 = *(const v16h*)(Bp + ((chunk * 4 + nt0 + 1) * 32 + lane) * 16);
    c0 = __builtin_amdgcn_wmma_f32_16x16x32_f16(false, a, false, b0, (short)0, c0, false, false);
    c1 = __builtin_amdgcn_wmma_f32_16x16x32_f16(false, a, false, b1, (short)0, c1, false, false);
  }

  // ---- epilogue: C layout -> VGPR j holds M = j + 8*(lane>=16), N = lane&15 ----
  const int mrow_base = (lane >> 4) * 8;
  const int ncol = lane & 15;
#pragma unroll
  for (int j = 0; j < 8; ++j) {
    const int node = node0 + mtile * 16 + mrow_base + j;
    if (node < N_NODES) {
      const int col0 = nt0 * 16 + ncol;
      out[(long)node * 64 + col0]      = c0[j] + bias[col0];
      out[(long)node * 64 + col0 + 16] = c1[j] + bias[col0 + 16];
    }
  }
}

extern "C" void kernel_launch(void* const* d_in, const int* in_sizes, int n_in,
                              void* d_out, int out_size, void* d_ws, size_t ws_size,
                              hipStream_t stream) {
  const float* x    = (const float*)d_in[0];
  const float* W    = (const float*)d_in[1];   // [7,64,64]
  const float* root = (const float*)d_in[2];   // [64,64]
  const float* bias = (const float*)d_in[3];   // [64]
  const int*   ei   = (const int*)d_in[4];     // [2,E]
  const int*   et   = (const int*)d_in[5];     // [E]
  float* out = (float*)d_out;

  float*    sums = (float*)d_ws;
  float*    cnt  = sums + SUMS_FLOATS;
  _Float16* Bp   = (_Float16*)(cnt + CNT_FLOATS);   // 32768 halves (64 KB)

  rgcn_zero<<<4096, 256, 0, stream>>>(sums, (long)(SUMS_FLOATS + CNT_FLOATS));
  rgcn_packB<<<dim3(4, 16), 32, 0, stream>>>(W, root, Bp);
  rgcn_scatter<<<N_EDGES / 8, 256, 0, stream>>>(x, ei, et, sums, cnt);
  rgcn_gemm<<<(N_NODES + 63) / 64, 256, 0, stream>>>(x, bias, sums, cnt, Bp, out);
}
